// ObjectContextBlock_48670569398403
// MI455X (gfx1250) — matrix operand order repaired and checked by reading, not compiled, not run on hardware
//
#include <hip/hip_runtime.h>
#include <hip/hip_bf16.h>

// ---------------------------------------------------------------------------
// ObjectContextBlock for MI455X (gfx1250): wave32, v_wmma_f32_16x16x32_f16,
// async global<->LDS DMA (global_load_async_to_lds_b128 /
// global_store_async_from_lds_b128, ASYNCcnt), double-buffered K pipeline,
// vectorized ds_load_b128 fragment gathers, fused BN(eval)+ReLU epilogues.
//
// Pipeline:
//   0) cvt weights wq1/wq2/wo -> f16 (tiny)
//   1) gemm<float,f16>  : q1  = relu(bn(x @ wq1^T))    (B,HW,256) f16
//   2) gemm<f16 ,f16>   : q2  = relu(bn(q1 @ wq2^T))   (B,HW,256) f16   [async A]
//   3) small_proj x3    : k,v proxy paths (152 rows)
//   4) attention        : WMMA sim/ctx, fp32 softmax over Kp=19 (pad 32),
//                         ctx drained via async store-from-LDS
//   5) gemm<f16 ,float> : out = relu(bn(ctx @ wo^T)) -> (B,C,H,W)       [async A]
// ---------------------------------------------------------------------------

typedef __attribute__((ext_vector_type(16))) _Float16 v16h;
typedef __attribute__((ext_vector_type(8)))  float    v8f;
typedef __attribute__((ext_vector_type(4)))  unsigned u32x4;
typedef __attribute__((ext_vector_type(4)))  float    f32x4;

#define INV_STD 0.9999950000374997f   // 1/sqrt(1+1e-5)

static __device__ __forceinline__ v8f wmma_f16(const v16h& a, const v16h& b, const v8f& c) {
    return __builtin_amdgcn_wmma_f32_16x16x32_f16(false, a, false, b, (short)0, c, false, false);
}

// Async DMA paths (ASYNCcnt): 16B global <-> LDS.
static __device__ __forceinline__ void async_ld_b128(unsigned lds_off, unsigned goff, const void* base) {
    asm volatile("global_load_async_to_lds_b128 %0, %1, %2"
                 :: "v"(lds_off), "v"(goff), "s"((unsigned long long)(uintptr_t)base)
                 : "memory");
}
static __device__ __forceinline__ void async_st_b128(unsigned goff, unsigned lds_off, void* base) {
    asm volatile("global_store_async_from_lds_b128 %0, %1, %2"
                 :: "v"(goff), "v"(lds_off), "s"((unsigned long long)(uintptr_t)base)
                 : "memory");
}
static __device__ __forceinline__ void wait_async0() {
    asm volatile("s_wait_asynccnt 0x0" ::: "memory");
}

// Fragment load: two 16B-aligned LDS chunks -> v16h (forces ds_load_b128 x2).
static __device__ __forceinline__ v16h ld_frag_pair(const _Float16* p0, const _Float16* p1) {
    union { u32x4 u[2]; v16h h; } f;
    f.u[0] = *(const u32x4*)p0;
    f.u[1] = *(const u32x4*)p1;
    return f.h;
}
// A(16x32) fragment: lane(hsel,ln) reads row halfs [8h..8h+7] and [16+8h..16+8h+7]
static __device__ __forceinline__ v16h ld_afrag(const _Float16* row, int hsel) {
    return ld_frag_pair(row + hsel * 8, row + 16 + hsel * 8);
}
// B(32x16) fragment: lane(hsel,n) reads row halfs [8h..8h+7] and [16+8h..16+8h+7]
static __device__ __forceinline__ v16h ld_bfrag(const _Float16* row, int hsel) {
    return ld_frag_pair(row + hsel * 8, row + 16 + hsel * 8);
}

// ---------------------------------------------------------------------------
// Tiled GEMM: Out[row,col] = relu((sum_k A[row,k]*Wh[col,k]) * INV_STD*g[col] + beta[col])
// Block 256 threads = 8 waves. Tile 256 rows x 128 cols, K stepped by 32,
// double-buffered LDS; wave owns 32 rows (2 A-frags) x 8 n-tiles.
// ---------------------------------------------------------------------------
template<typename AT, typename OT>
__global__ __launch_bounds__(256)
void gemm_bn_relu_kernel(const AT* __restrict__ A, const _Float16* __restrict__ Wh,
                         const float* __restrict__ gamma, const float* __restrict__ beta,
                         OT* __restrict__ Out, int K, int a_async,
                         long long a_batch, long long a_rs, long long a_ks,
                         long long o_batch, long long o_rs, long long o_cs)
{
    __shared__ __align__(16) _Float16 As[2][256][40];  // 80B rows: 16B aligned, low-conflict
    __shared__ __align__(16) _Float16 Ws[2][128][40];

    const int tid  = threadIdx.x;
    const int lane = tid & 31;
    const int wave = tid >> 5;
    const int hsel = lane >> 4;
    const int ln   = lane & 15;
    const long long row0 = (long long)blockIdx.x * 256;
    const int       col0 = blockIdx.y * 128;

    const AT* Ab = A + (long long)blockIdx.z * a_batch;
    OT*       Ob = Out + (long long)blockIdx.z * o_batch;

    v8f acc0[8], acc1[8];
#pragma unroll
    for (int i = 0; i < 8; i++) {
        acc0[i] = (v8f){0.f,0.f,0.f,0.f,0.f,0.f,0.f,0.f};
        acc1[i] = acc0[i];
    }

    auto stage = [&](int kt, int buf) {
        if (a_async) {
            // A tile: 256 rows x 64B -> 1024 x b128, 4 per thread (ASYNCcnt)
#pragma unroll
            for (int j = 0; j < 4; j++) {
                const int idx = tid + 256 * j;
                const int m = idx >> 2, c = idx & 3;
                const unsigned lds = (unsigned)(uintptr_t)&As[buf][m][c * 8];
                const unsigned go  = (unsigned)(((row0 + m) * a_rs + kt) * 2) + c * 16;
                async_ld_b128(lds, go, Ab);
            }
        } else {
            // f32 strided A (x): thread = one row, 32 coalesced loads + cvt
            const int m = tid;
#pragma unroll
            for (int kk = 0; kk < 32; kk++)
                As[buf][m][kk] = (_Float16)(float)Ab[(row0 + m) * a_rs + (long long)(kt + kk) * a_ks];
        }
        // W tile: 128 rows x 64B -> 512 x b128, 2 per thread (ASYNCcnt)
#pragma unroll
        for (int j = 0; j < 2; j++) {
            const int idx = tid + 256 * j;
            const int n = idx >> 2, c = idx & 3;
            const unsigned lds = (unsigned)(uintptr_t)&Ws[buf][n][c * 8];
            const unsigned go  = (unsigned)(((long long)(col0 + n) * K + kt) * 2) + c * 16;
            async_ld_b128(lds, go, Wh);
        }
    };

    stage(0, 0);
    wait_async0();
    __syncthreads();

    int buf = 0;
    for (int kt = 0; kt < K; kt += 32) {
        if (kt + 32 < K) stage(kt + 32, buf ^ 1);   // prefetch overlaps compute

        const int m0 = wave * 32 + ln;
        const v16h a0 = ld_afrag(&As[buf][m0][0],      hsel);
        const v16h a1 = ld_afrag(&As[buf][m0 + 16][0], hsel);
#pragma unroll
        for (int nt = 0; nt < 8; nt++) {
            const v16h bf = ld_bfrag(&Ws[buf][nt * 16 + ln][0], hsel);
            acc0[nt] = wmma_f16(a0, bf, acc0[nt]);   // one B-frag -> two WMMAs
            acc1[nt] = wmma_f16(a1, bf, acc1[nt]);
        }
        wait_async0();
        __syncthreads();
        buf ^= 1;
    }

    // fused BN(eval)+ReLU epilogue; C/D layout: lane -> (M = r + 8*hsel, N = ln)
#pragma unroll
    for (int nt = 0; nt < 8; nt++) {
        const int col = col0 + nt * 16 + ln;
        const float sc = INV_STD * gamma[col];
        const float bi = beta[col];
        if constexpr (sizeof(OT) == 4) {
            // o_rs==1 (channel-major output): 8 rows per lane are contiguous f32
            float t0[8], t1[8];
#pragma unroll
            for (int r = 0; r < 8; r++) {
                float v0 = acc0[nt][r] * sc + bi;  t0[r] = v0 > 0.f ? v0 : 0.f;
                float v1 = acc1[nt][r] * sc + bi;  t1[r] = v1 > 0.f ? v1 : 0.f;
            }
            const long long base = row0 + wave * 32 + 8 * hsel;
            float* p0 = (float*)Ob + base * o_rs + (long long)col * o_cs;
            float* p1 = (float*)Ob + (base + 16) * o_rs + (long long)col * o_cs;
            *(f32x4*)(p0)     = *(f32x4*)&t0[0];
            *(f32x4*)(p0 + 4) = *(f32x4*)&t0[4];
            *(f32x4*)(p1)     = *(f32x4*)&t1[0];
            *(f32x4*)(p1 + 4) = *(f32x4*)&t1[4];
        } else {
#pragma unroll
            for (int r = 0; r < 8; r++) {
                const long long rowA = row0 + wave * 32 + r + 8 * hsel;
                float v0 = acc0[nt][r] * sc + bi;  v0 = v0 > 0.f ? v0 : 0.f;
                float v1 = acc1[nt][r] * sc + bi;  v1 = v1 > 0.f ? v1 : 0.f;
                Ob[rowA * o_rs + (long long)col * o_cs]        = (OT)v0;
                Ob[(rowA + 16) * o_rs + (long long)col * o_cs] = (OT)v1;
            }
        }
    }
}

// ---------------------------------------------------------------------------
__global__ __launch_bounds__(256)
void cvt_f16_kernel(const float* __restrict__ src, _Float16* __restrict__ dst, int n)
{
    const int i = blockIdx.x * 256 + threadIdx.x;
    if (i < n) dst[i] = (_Float16)src[i];
}

// ---------------------------------------------------------------------------
// Tiny proxy projections (rows = B*Kp = 152): one thread per output element.
// ---------------------------------------------------------------------------
__global__ __launch_bounds__(256)
void small_proj_kernel(const float* __restrict__ A, const float* __restrict__ W,
                       const float* __restrict__ gamma, const float* __restrict__ beta,
                       float* __restrict__ Out,
                       int rows, int K, int O, int Kp, int a_mode, int o_mode)
{
    const int idx = blockIdx.x * 256 + threadIdx.x;
    if (idx >= rows * O) return;
    const int row = idx / O;
    const int o   = idx - row * O;
    const int bb  = row / Kp;
    const int p   = row - bb * Kp;
    float s = 0.f;
    for (int k = 0; k < K; k++) {
        const float a = (a_mode == 0) ? A[((long long)bb * K + k) * Kp + p]
                                      : A[(long long)row * K + k];
        s += a * W[(long long)o * K + k];
    }
    float val = s * (INV_STD * gamma[o]) + beta[o];
    val = val > 0.f ? val : 0.f;
    if (o_mode == 0) Out[(long long)row * O + o]           = val;
    else             Out[((long long)bb * O + o) * Kp + p] = val;
}

// ---------------------------------------------------------------------------
// Attention: per block = 128 pixels of one batch.
//   sim(16x19) = q(16x256) @ k(256x19)  [19 padded to 32 -> 2 WMMA n-tiles]
//   fp32 softmax over 19 (shfl_xor within 16-lane groups)
//   ctx(16x256) = att(16x32) @ v^T(32x256) [1 K-step, 16 WMMA n-tiles]
// q tile in via async DMA; ctx tiles restaged into Qs (dead after sim) and
// drained with global_store_async_from_lds_b128 (coalesced 512B rows).
// ---------------------------------------------------------------------------
__global__ __launch_bounds__(256)
void attention_kernel(const _Float16* __restrict__ q,      // (B,HW,256) f16
                      const float* __restrict__ kmat,      // (B,256,19)
                      const float* __restrict__ vmat,      // (B,256,19)
                      _Float16* __restrict__ ctx,          // (B,HW,256) f16
                      int HW)
{
    __shared__ __align__(16) _Float16 Qs[128][264];   // 528B rows (16B aligned)
    __shared__ __align__(16) _Float16 Ks[32][264];    // [p][ch], rows p>=19 zeroed
    __shared__ __align__(16) _Float16 Vs[256][40];    // [ch][p], cols p>=19 zeroed
    __shared__ __align__(16) _Float16 AttS[8][16][40];

    const int tid  = threadIdx.x;
    const int lane = tid & 31;
    const int wave = tid >> 5;
    const int hsel = lane >> 4;
    const int ln   = lane & 15;
    const int b    = blockIdx.y;
    const long long pix0 = (long long)blockIdx.x * 128;

    // q tile: 128 rows x 512B -> 4096 x b128 async, 16 per thread
#pragma unroll
    for (int j = 0; j < 16; j++) {
        const int idx = tid + 256 * j;
        const int m = idx >> 5, c = idx & 31;
        const unsigned lds = (unsigned)(uintptr_t)&Qs[m][c * 8];
        const unsigned go  = (unsigned)((((long long)b * HW + pix0 + m) * 256 + c * 8) * 2);
        async_ld_b128(lds, go, q);
    }
    // K (transpose + zero-pad p>=19)
    for (int idx = tid; idx < 32 * 256; idx += 256) {
        const int p  = idx & 31;
        const int ch = idx >> 5;
        const float v = (p < 19) ? kmat[(long long)b * 256 * 19 + ch * 19 + p] : 0.f;
        Ks[p][ch] = (_Float16)v;
    }
    // V (zero-pad p>=19)
    for (int idx = tid; idx < 256 * 32; idx += 256) {
        const int p  = idx & 31;
        const int ch = idx >> 5;
        const float v = (p < 19) ? vmat[(long long)b * 256 * 19 + ch * 19 + p] : 0.f;
        Vs[ch][p] = (_Float16)v;
    }
    wait_async0();
    __syncthreads();

    // ---- sim = q @ k over K=256 (8 WMMA steps x 2 n-tiles) ----
    v8f s0 = (v8f){0.f,0.f,0.f,0.f,0.f,0.f,0.f,0.f};
    v8f s1 = s0;
#pragma unroll
    for (int kt = 0; kt < 256; kt += 32) {
        const v16h afrag = ld_afrag(&Qs[wave * 16 + ln][kt], hsel);
        const v16h b0    = ld_bfrag(&Ks[ln][kt],      hsel);
        const v16h b1    = ld_bfrag(&Ks[16 + ln][kt], hsel);
        s0 = wmma_f16(afrag, b0, s0);
        s1 = wmma_f16(afrag, b1, s1);
    }

    // ---- fp32 softmax over 19 columns ----
    const float scale = 0.0625f;             // 256^-0.5
    const bool valid1 = (16 + ln) < 19;
    float e0[8], e1[8];
#pragma unroll
    for (int r = 0; r < 8; r++) {
        const float v0 = s0[r] * scale;
        const float v1 = valid1 ? s1[r] * scale : -3.0e38f;
        float mx = fmaxf(v0, v1);
#pragma unroll
        for (int off = 8; off >= 1; off >>= 1) mx = fmaxf(mx, __shfl_xor(mx, off, 32));
        const float x0 = __expf(v0 - mx);
        const float x1 = valid1 ? __expf(v1 - mx) : 0.f;
        float sm = x0 + x1;
#pragma unroll
        for (int off = 8; off >= 1; off >>= 1) sm += __shfl_xor(sm, off, 32);
        const float inv = 1.f / sm;
        e0[r] = x0 * inv;
        e1[r] = x1 * inv;
    }
    // restage att (C-layout -> LDS); barrier also guarantees all waves are done
    // reading Qs (sim) before we overwrite it with ctx below.
#pragma unroll
    for (int r = 0; r < 8; r++) {
        const int row = r + 8 * hsel;
        AttS[wave][row][ln]      = (_Float16)e0[r];
        AttS[wave][row][16 + ln] = (_Float16)e1[r];
    }
    __syncthreads();

    const v16h attfrag = ld_afrag(&AttS[wave][ln][0], hsel);

    // ---- ctx = att @ v^T (single zero-padded K step, 16 n-tiles) ----
#pragma unroll
    for (int nt = 0; nt < 16; nt++) {
        const int ch = nt * 16 + ln;
        const v16h bfrag = ld_bfrag(&Vs[ch][0], hsel);
        v8f c = (v8f){0.f,0.f,0.f,0.f,0.f,0.f,0.f,0.f};
        c = wmma_f16(attfrag, bfrag, c);
#pragma unroll
        for (int r = 0; r < 8; r++)
            Qs[wave * 16 + r + 8 * hsel][ch] = (_Float16)c[r];   // ctx staging in Qs
    }
    __syncthreads();

    // drain ctx: 128 rows x 512B via async store-from-LDS (s_endpgm waits idle)
#pragma unroll
    for (int j = 0; j < 16; j++) {
        const int idx = tid + 256 * j;
        const int m = idx >> 5, c = idx & 31;
        const unsigned lds = (unsigned)(uintptr_t)&Qs[m][c * 8];
        const unsigned go  = (unsigned)((((long long)b * HW + pix0 + m) * 256 + c * 8) * 2);
        async_st_b128(go, lds, ctx);
    }
    wait_async0();
}

// ---------------------------------------------------------------------------
extern "C" void kernel_launch(void* const* d_in, const int* in_sizes, int n_in,
                              void* d_out, int out_size, void* d_ws, size_t ws_size,
                              hipStream_t stream) {
    const float* x     = (const float*)d_in[0];
    const float* proxy = (const float*)d_in[1];
    const float* wq1   = (const float*)d_in[2];
    const float* gq1   = (const float*)d_in[3];
    const float* bq1   = (const float*)d_in[4];
    const float* wq2   = (const float*)d_in[5];
    const float* gq2   = (const float*)d_in[6];
    const float* bq2   = (const float*)d_in[7];
    const float* wk1   = (const float*)d_in[8];
    const float* gk1   = (const float*)d_in[9];
    const float* bk1   = (const float*)d_in[10];
    const float* wk2   = (const float*)d_in[11];
    const float* gk2   = (const float*)d_in[12];
    const float* bk2   = (const float*)d_in[13];
    const float* wv    = (const float*)d_in[14];
    const float* gv    = (const float*)d_in[15];
    const float* bv    = (const float*)d_in[16];
    const float* wo    = (const float*)d_in[17];
    const float* go    = (const float*)d_in[18];
    const float* bo    = (const float*)d_in[19];
    float* out = (float*)d_out;

    const int B = 8, C = 512, HW = 128 * 128, Kc = 256, Kp = 19;

    // workspace layout (q1 region reused for ctx after q2 is produced)
    _Float16* q1   = (_Float16*)d_ws;
    _Float16* q2   = q1 + (size_t)B * HW * Kc;
    _Float16* ctx  = q1;
    float*    k1   = (float*)(q2 + (size_t)B * HW * Kc);
    float*    kf   = k1 + (size_t)B * Kp * Kc;
    float*    vf   = kf + (size_t)B * Kp * Kc;
    _Float16* wq1h = (_Float16*)(vf + (size_t)B * Kp * Kc);
    _Float16* wq2h = wq1h + (size_t)Kc * C;
    _Float16* woh  = wq2h + (size_t)Kc * Kc;

    const dim3 blk(256);

    // 0) pre-convert big weights to f16 (enables pure async-DMA W staging)
    cvt_f16_kernel<<<(Kc * C + 255) / 256, blk, 0, stream>>>(wq1, wq1h, Kc * C);
    cvt_f16_kernel<<<(Kc * Kc + 255) / 256, blk, 0, stream>>>(wq2, wq2h, Kc * Kc);
    cvt_f16_kernel<<<(C * Kc + 255) / 256, blk, 0, stream>>>(wo, woh, C * Kc);

    // 1) q1 = relu(bn(x @ wq1^T)) : A = x (B,C,HW): row stride 1, k stride HW (sync A)
    gemm_bn_relu_kernel<float, _Float16><<<dim3(HW / 256, Kc / 128, B), blk, 0, stream>>>(
        x, wq1h, gq1, bq1, q1, C, /*a_async=*/0,
        (long long)C * HW, 1LL, (long long)HW,
        (long long)HW * Kc, (long long)Kc, 1LL);

    // 2) q2 = relu(bn(q1 @ wq2^T)) : A row-major f16 -> async A staging
    gemm_bn_relu_kernel<_Float16, _Float16><<<dim3(HW / 256, Kc / 128, B), blk, 0, stream>>>(
        q1, wq2h, gq2, bq2, q2, Kc, /*a_async=*/1,
        (long long)HW * Kc, (long long)Kc, 1LL,
        (long long)HW * Kc, (long long)Kc, 1LL);

    // 3) proxy k/v paths (152 rows total -- trivial)
    const int rows = B * Kp;
    const int tot  = rows * Kc;
    small_proj_kernel<<<(tot + 255) / 256, blk, 0, stream>>>(proxy, wk1, gk1, bk1, k1, rows, C,  Kc, Kp, 0, 0);
    small_proj_kernel<<<(tot + 255) / 256, blk, 0, stream>>>(k1,    wk2, gk2, bk2, kf, rows, Kc, Kc, Kp, 1, 1);
    small_proj_kernel<<<(tot + 255) / 256, blk, 0, stream>>>(proxy, wv,  gv,  bv,  vf, rows, C,  Kc, Kp, 0, 1);

    // 4) attention (writes ctx into q1's region; q1 is dead by now)
    attention_kernel<<<dim3(HW / 128, B), blk, 0, stream>>>(q2, kf, vf, ctx, HW);

    // 5) out = relu(bn(ctx @ wo^T)) -> (B,C,H,W): row(hw) stride 1, col(c) stride HW
    gemm_bn_relu_kernel<_Float16, float><<<dim3(HW / 256, C / 128, B), blk, 0, stream>>>(
        ctx, woh, go, bo, out, Kc, /*a_async=*/1,
        (long long)HW * Kc, (long long)Kc, 1LL,
        (long long)C * HW, 1LL, (long long)HW);
}
